// GRUNetv4_extra_attn_48284022342320
// MI455X (gfx1250) — matrix operand order repaired and checked by reading, not compiled, not run on hardware
//
#include <hip/hip_runtime.h>
#include <hip/hip_bf16.h>

// ---------------------------------------------------------------------------
// GRUNet (BN -> GRU -> identity-attn -> 3xMLP) for MI455X / gfx1250.
// Strategy: bf16 WMMA (v_wmma_f32_16x16x32_bf16) for every GEMM, f32 gate math.
//   1) BN stats + normalize-to-bf16
//   2) Gi = xn @ Wi^T + (bi+bh)  -- hoists all time-parallel GEMM work
//   3) 512 serial steps: Gh = h @ Wh^T (WMMA), fused gate kernel
//   4) MLP: 3 WMMA GEMMs with ReLU (attn weight == 1, identity -> skipped)
// Tensor Data Mover (tensor_load_to_lds, 6-arg clang-23 form) with
// double-buffered LDS tiles: the global->LDS DMA of tile k+1 overlaps WMMA
// on tile k, synchronized with s_wait_tensorcnt + workgroup barriers.
// ---------------------------------------------------------------------------

#define T_LEN 512
#define B_SZ  256
#define I_SZ  128
#define H_SZ  256
#define H3    (3 * H_SZ)

typedef __attribute__((ext_vector_type(16))) __bf16        v16bf;
typedef __attribute__((ext_vector_type(8)))  float         v8f;
typedef __attribute__((ext_vector_type(8)))  unsigned int  v8u;
typedef __attribute__((ext_vector_type(4)))  unsigned int  v4u;
typedef __attribute__((ext_vector_type(8)))  int           v8i;
typedef __attribute__((ext_vector_type(4)))  int           v4i;

#if __has_builtin(__builtin_amdgcn_tensor_load_to_lds)
#define USE_TDM 1
#else
#define USE_TDM 0
#endif

__device__ __forceinline__ unsigned short f2bf(float f) {
    unsigned int u = __float_as_uint(f);
    unsigned int r = u + 0x7FFFu + ((u >> 16) & 1u);   // round-to-nearest-even
    return (unsigned short)(r >> 16);
}

#if USE_TDM
// ---------------------------------------------------------------------------
// Issue one TDM descriptor: DMA a [128 rows x 32 cols] bf16 tile from a
// row-major [rows_total x K] tensor (tile starts at gaddr) into LDS at
// lds_addr, padding each 64B row with 8B -> LDS row stride 72B (= LDA 36).
// D# layout per CDNA5 ISA 8.3/8.4; issued by one wave, tracked by TENSORcnt.
// Toolchain builtin is the 6-arg form: (g0, g1, g2, g3, g_extra, cpol).
// ---------------------------------------------------------------------------
__device__ __forceinline__ void tdm_load_tile(const unsigned short* gaddr,
                                              unsigned lds_addr,
                                              int K_total, int rows_total) {
    unsigned long long ga = (unsigned long long)(uintptr_t)gaddr;
    v4u g0;
    g0[0] = 1u;                                            // count=1, user mode
    g0[1] = lds_addr;                                      // LDS byte address
    g0[2] = (unsigned)(ga & 0xFFFFFFFFu);                  // global_addr[31:0]
    g0[3] = (unsigned)((ga >> 32) & 0x01FFFFFFu)           // global_addr[56:32]
            | (2u << 30);                                  // type=2 ("image")
    v8i g1;
    g1[0] = (int)((1u << 16)      // data_size = 2 bytes
                | (1u << 20)      // pad_enable
                | (3u << 22)      // pad_interval: 16 DWORDs (one 32-elt row)
                | (1u << 25));    // pad_amount: 2 DWORDs (8B -> stride 72B)
    g1[1] = (int)(((unsigned)K_total & 0xFFFFu) << 16);    // tensor_dim0 lo
    g1[2] = (int)(((unsigned)K_total >> 16)                // tensor_dim0 hi
                | (((unsigned)rows_total & 0xFFFFu) << 16)); // tensor_dim1 lo
    g1[3] = (int)((((unsigned)rows_total >> 16) & 0xFFFFu) // tensor_dim1 hi
                | (32u << 16));                            // tile_dim0 = 32
    g1[4] = (int)128u;                                     // tile_dim1 = 128
    g1[5] = (int)K_total;                                  // dim0_stride lo
    g1[6] = 0;                                             // dim0_stride hi
    g1[7] = 0;
    v4i z4 = {0, 0, 0, 0};                                 // 2D: groups 2/3 zero
    v8i z8 = {0, 0, 0, 0, 0, 0, 0, 0};
    __builtin_amdgcn_tensor_load_to_lds(g0, g1, z4, z4, z8, 0);
}
#endif

// ---------------------------------------------------------------------------
// BatchNorm statistics: one block per feature (I=128 blocks).
// ---------------------------------------------------------------------------
__global__ __launch_bounds__(256)
void bn_stats_kernel(const float* __restrict__ x,
                     float* __restrict__ mean, float* __restrict__ rstd) {
    __shared__ float s1[256], s2[256];
    const int i = blockIdx.x, tid = threadIdx.x;
    float a = 0.f, b = 0.f;
    for (int r = tid; r < T_LEN * B_SZ; r += 256) {
        float v = x[(size_t)r * I_SZ + i];
        a += v; b += v * v;
    }
    s1[tid] = a; s2[tid] = b;
    __syncthreads();
    for (int s = 128; s > 0; s >>= 1) {
        if (tid < s) { s1[tid] += s1[tid + s]; s2[tid] += s2[tid + s]; }
        __syncthreads();
    }
    if (tid == 0) {
        const float inv_n = 1.0f / (float)(T_LEN * B_SZ);
        float mu  = s1[0] * inv_n;
        float var = s2[0] * inv_n - mu * mu;            // biased variance
        mean[i] = mu;
        rstd[i] = rsqrtf(var + 1e-5f);
    }
}

// Normalize x and emit bf16 xn.
__global__ __launch_bounds__(256)
void bn_apply_kernel(const float* __restrict__ x,
                     const float* __restrict__ mean, const float* __restrict__ rstd,
                     const float* __restrict__ gamma, const float* __restrict__ beta,
                     unsigned short* __restrict__ xn) {
    int idx = blockIdx.x * 256 + threadIdx.x;
    if (idx >= T_LEN * B_SZ * I_SZ) return;
    int i = idx & (I_SZ - 1);
    float v = (x[idx] - mean[i]) * rstd[i] * gamma[i] + beta[i];
    xn[idx] = f2bf(v);
}

// Generic f32 -> bf16 weight conversion.
__global__ __launch_bounds__(256)
void cvt_bf16_kernel(const float* __restrict__ src,
                     unsigned short* __restrict__ dst, int n) {
    int idx = blockIdx.x * 256 + threadIdx.x;
    if (idx < n) dst[idx] = f2bf(src[idx]);
}

// Init hidden state from input h, and fold the two GRU biases.
__global__ __launch_bounds__(256)
void init_kernel(const float* __restrict__ h_in,
                 const float* __restrict__ bi, const float* __restrict__ bh,
                 float* __restrict__ h_f32, unsigned short* __restrict__ h_bf,
                 float* __restrict__ bias_c) {
    int idx = blockIdx.x * 256 + threadIdx.x;
    if (idx < B_SZ * H_SZ) {
        float v = h_in[idx];
        h_f32[idx] = v;
        h_bf[idx]  = f2bf(v);
    }
    if (idx < H3) bias_c[idx] = bi[idx] + bh[idx];
}

// ---------------------------------------------------------------------------
// Tiled bf16 WMMA GEMM:  out[M,N] = A[M,K] @ Wt[N,K]^T (+ bias) (+ ReLU)
// 256 threads = 8 waves (wave32). Block tile 128x128, K staged in 32-chunks,
// double-buffered in LDS. With TDM: wave0/wave1 DMA the next A/W tiles while
// all waves run WMMA on the current tiles (s_wait_tensorcnt <= 1 pipelining).
// Fragment packing per CDNA5 ISA 7.12.2:
//   A 16x32 bf16 : lane -> row M=lane&15; K pairs, lane-half selects K base 0/8.
//   B 32x16 bf16 : lane -> col N=lane&15; K pairs, lane-half selects K base 0/8.
//   C/D 16x16 f32: VGPR r -> row M = r + (lane<16 ? 0 : 8), col N = lane&15.
// ---------------------------------------------------------------------------
template <bool BIAS, bool RELU, bool OUT_BF16>
__global__ __launch_bounds__(256)
void gemm_bf16_wmma(const unsigned short* __restrict__ A,
                    const unsigned short* __restrict__ Wt,   // [N][K] row-major
                    const float* __restrict__ bias,
                    void* __restrict__ out,
                    int M, int N, int K) {
    constexpr int BM = 128, BN = 128, KC = 32, LDA = 36;     // LDA pad: bank spread
    __shared__ __align__(16) unsigned short sA[2][BM * LDA];
    __shared__ __align__(16) unsigned short sW[2][BN * LDA];

    const int tid  = threadIdx.x;
    const int lane = tid & 31;
    const int wid  = tid >> 5;
    const int wm   = wid >> 2;           // 0..1  (64 rows each)
    const int wn   = wid & 3;            // 0..3  (32 cols each)
    const int bm0  = blockIdx.x * BM;
    const int bn0  = blockIdx.y * BN;

    v8f acc[4][2];
    for (int i = 0; i < 4; ++i)
        for (int j = 0; j < 2; ++j)
            for (int r = 0; r < 8; ++r) acc[i][j][r] = 0.f;

    const int m_l  = lane & 15;
    const int a_kb = (lane < 16) ? 0 : 8;
    const int n_l  = lane & 15;
    const int b_kb = (lane < 16) ? 0 : 8;
    const int NK   = K / KC;

#if USE_TDM
    // Prologue: DMA first tiles into buffer 0.
    if (wid == 0)
        tdm_load_tile(A  + (size_t)bm0 * K, (unsigned)(uintptr_t)&sA[0][0], K, M);
    if (wid == 1)
        tdm_load_tile(Wt + (size_t)bn0 * K, (unsigned)(uintptr_t)&sW[0][0], K, N);
#endif

    for (int c = 0; c < NK; ++c) {
        const int k0  = c * KC;
        const int buf = c & 1;
#if USE_TDM
        const bool more = (c + 1 < NK);
        if (more) {  // prefetch next K tile into the other buffer
            if (wid == 0)
                tdm_load_tile(A  + (size_t)bm0 * K + (k0 + KC),
                              (unsigned)(uintptr_t)&sA[buf ^ 1][0], K, M);
            if (wid == 1)
                tdm_load_tile(Wt + (size_t)bn0 * K + (k0 + KC),
                              (unsigned)(uintptr_t)&sW[buf ^ 1][0], K, N);
        }
        if (wid < 2) {  // wait for the CURRENT tile only (older of the 2 DMAs)
            if (more) __builtin_amdgcn_s_wait_tensorcnt(1);
            else      __builtin_amdgcn_s_wait_tensorcnt(0);
        }
        __syncthreads();
#else
        // Fallback: cooperative synchronous staging.
        for (int i = tid; i < BM * (KC / 4); i += 256) {
            int row = i >> 3, c4 = (i & 7) * 4;
            *(unsigned long long*)&sA[buf][row * LDA + c4] =
                *(const unsigned long long*)&A[(size_t)(bm0 + row) * K + k0 + c4];
        }
        for (int i = tid; i < BN * (KC / 4); i += 256) {
            int row = i >> 3, c4 = (i & 7) * 4;
            *(unsigned long long*)&sW[buf][row * LDA + c4] =
                *(const unsigned long long*)&Wt[(size_t)(bn0 + row) * K + k0 + c4];
        }
        __syncthreads();
#endif

        v8u bfrag[2];
        for (int tn = 0; tn < 2; ++tn)
            for (int j = 0; j < 8; ++j)
                bfrag[tn][j] = *(const unsigned int*)
                    &sW[buf][(wn * 32 + tn * 16 + n_l) * LDA + b_kb + ((j < 4) ? 2 * j : 16 + 2 * (j - 4))];

        for (int tm = 0; tm < 4; ++tm) {
            v8u afrag;
            for (int j = 0; j < 8; ++j) {
                int kk = a_kb + ((j < 4) ? 2 * j : 16 + 2 * (j - 4));
                afrag[j] = *(const unsigned int*)
                    &sA[buf][(wm * 64 + tm * 16 + m_l) * LDA + kk];
            }
            for (int tn = 0; tn < 2; ++tn) {
                acc[tm][tn] = __builtin_amdgcn_wmma_f32_16x16x32_bf16(
                    false, __builtin_bit_cast(v16bf, afrag),
                    false, __builtin_bit_cast(v16bf, bfrag[tn]),
                    (short)0, acc[tm][tn], false, false);
            }
        }
        __syncthreads();   // all reads of this buffer done before it's re-DMA'd
    }

    const int mrow = (lane < 16) ? 0 : 8;
    for (int tm = 0; tm < 4; ++tm) {
        for (int tn = 0; tn < 2; ++tn) {
            int col = bn0 + wn * 32 + tn * 16 + n_l;
            float bv = 0.f;
            if constexpr (BIAS) bv = bias[col];
            for (int r = 0; r < 8; ++r) {
                int row = bm0 + wm * 64 + tm * 16 + mrow + r;
                float v = acc[tm][tn][r] + bv;
                if constexpr (RELU) v = v > 0.f ? v : 0.f;
                if constexpr (OUT_BF16)
                    ((unsigned short*)out)[(size_t)row * N + col] = f2bf(v);
                else
                    ((float*)out)[(size_t)row * N + col] = v;
            }
        }
    }
}

// ---------------------------------------------------------------------------
// GRU gate math for one timestep (biases already folded into Gi).
// Also emits h as bf16 (next step's GEMM A) and outs*attn_w (==1) as bf16.
// Prefetches the next timestep's Gi slice into L2 (global_prefetch_b8) to
// shorten the serial recurrence chain.
// ---------------------------------------------------------------------------
__global__ __launch_bounds__(256)
void gru_gate_kernel(int t,
                     const float* __restrict__ Gi,   // [T][B][3H]
                     const float* __restrict__ Gh,   // [B][3H]
                     float* __restrict__ h,          // [B][H] f32
                     unsigned short* __restrict__ h_bf,
                     unsigned short* __restrict__ outs_bf,
                     float* __restrict__ hlast_out) {
    int idx = blockIdx.x * 256 + threadIdx.x;
    if (idx >= B_SZ * H_SZ) return;
    int b = idx >> 8;                     // H = 256
    int c = idx & (H_SZ - 1);
    const float* gi = Gi + ((size_t)t * B_SZ + b) * H3;
    const float* gh = Gh + (size_t)b * H3;

    if (t + 1 < T_LEN) {   // warm L2 for the next step of the serial chain
        __builtin_prefetch(gi + (size_t)B_SZ * H3 + c, 0, 1);
        __builtin_prefetch(gi + (size_t)B_SZ * H3 + H_SZ + c, 0, 1);
        __builtin_prefetch(gi + (size_t)B_SZ * H3 + 2 * H_SZ + c, 0, 1);
    }

    float r = 1.f / (1.f + __expf(-(gi[c]            + gh[c])));
    float z = 1.f / (1.f + __expf(-(gi[H_SZ + c]     + gh[H_SZ + c])));
    float n = tanhf(gi[2 * H_SZ + c] + r * gh[2 * H_SZ + c]);
    float hp = h[idx];
    float hn = (1.f - z) * n + z * hp;

    h[idx]    = hn;
    h_bf[idx] = f2bf(hn);
    outs_bf[(size_t)t * B_SZ * H_SZ + idx] = f2bf(hn);  // attn weight == 1
    if (t == T_LEN - 1) hlast_out[idx] = hn;
}

// ---------------------------------------------------------------------------
// Host-side orchestration.
// ---------------------------------------------------------------------------
static inline size_t ws_take(size_t& off, size_t bytes) {
    size_t a = off;
    off += (bytes + 255) & ~(size_t)255;
    return a;
}

extern "C" void kernel_launch(void* const* d_in, const int* in_sizes, int n_in,
                              void* d_out, int out_size, void* d_ws, size_t ws_size,
                              hipStream_t stream) {
    (void)in_sizes; (void)n_in; (void)out_size; (void)ws_size;

    const float* x      = (const float*)d_in[0];
    const float* h_in   = (const float*)d_in[1];
    const float* gamma  = (const float*)d_in[2];
    const float* beta   = (const float*)d_in[3];
    const float* gru_Wi = (const float*)d_in[4];
    const float* gru_Wh = (const float*)d_in[5];
    const float* gru_bi = (const float*)d_in[6];
    const float* gru_bh = (const float*)d_in[7];
    // d_in[8]/d_in[9]: attn_W / attn_b -- mathematically identity, unused.
    const float* p1_W = (const float*)d_in[10];
    const float* p1_b = (const float*)d_in[11];
    const float* p2_W = (const float*)d_in[12];
    const float* p2_b = (const float*)d_in[13];
    const float* p3_W = (const float*)d_in[14];
    const float* p3_b = (const float*)d_in[15];

    char* ws = (char*)d_ws;
    size_t off = 0;
    float*          mean    = (float*)(ws + ws_take(off, I_SZ * 4));
    float*          rstd    = (float*)(ws + ws_take(off, I_SZ * 4));
    float*          bias_c  = (float*)(ws + ws_take(off, H3 * 4));
    unsigned short* xn_bf   = (unsigned short*)(ws + ws_take(off, (size_t)T_LEN * B_SZ * I_SZ * 2));
    unsigned short* Wi_bf   = (unsigned short*)(ws + ws_take(off, (size_t)H3 * I_SZ * 2));
    unsigned short* Wh_bf   = (unsigned short*)(ws + ws_take(off, (size_t)H3 * H_SZ * 2));
    unsigned short* p1_bf   = (unsigned short*)(ws + ws_take(off, (size_t)H_SZ * H_SZ * 2));
    unsigned short* p2_bf   = (unsigned short*)(ws + ws_take(off, (size_t)H_SZ * H_SZ * 2));
    unsigned short* p3_bf   = (unsigned short*)(ws + ws_take(off, (size_t)H_SZ * H_SZ * 2));
    float*          Gi      = (float*)(ws + ws_take(off, (size_t)T_LEN * B_SZ * H3 * 4));
    float*          Gh      = (float*)(ws + ws_take(off, (size_t)B_SZ * H3 * 4));
    float*          h_f32   = (float*)(ws + ws_take(off, (size_t)B_SZ * H_SZ * 4));
    unsigned short* h_bf    = (unsigned short*)(ws + ws_take(off, (size_t)B_SZ * H_SZ * 2));
    unsigned short* outs_bf = (unsigned short*)(ws + ws_take(off, (size_t)T_LEN * B_SZ * H_SZ * 2));
    unsigned short* y1_bf   = (unsigned short*)(ws + ws_take(off, (size_t)T_LEN * B_SZ * H_SZ * 2));
    unsigned short* y2_bf   = (unsigned short*)(ws + ws_take(off, (size_t)T_LEN * B_SZ * H_SZ * 2));

    float* y_out     = (float*)d_out;                               // [T*B*H]
    float* hlast_out = (float*)d_out + (size_t)T_LEN * B_SZ * H_SZ; // [B*H]

    // 1) BatchNorm
    bn_stats_kernel<<<I_SZ, 256, 0, stream>>>(x, mean, rstd);
    bn_apply_kernel<<<(T_LEN * B_SZ * I_SZ + 255) / 256, 256, 0, stream>>>(
        x, mean, rstd, gamma, beta, xn_bf);

    // Weight conversions to bf16
    cvt_bf16_kernel<<<(H3 * I_SZ + 255) / 256, 256, 0, stream>>>(gru_Wi, Wi_bf, H3 * I_SZ);
    cvt_bf16_kernel<<<(H3 * H_SZ + 255) / 256, 256, 0, stream>>>(gru_Wh, Wh_bf, H3 * H_SZ);
    cvt_bf16_kernel<<<(H_SZ * H_SZ + 255) / 256, 256, 0, stream>>>(p1_W, p1_bf, H_SZ * H_SZ);
    cvt_bf16_kernel<<<(H_SZ * H_SZ + 255) / 256, 256, 0, stream>>>(p2_W, p2_bf, H_SZ * H_SZ);
    cvt_bf16_kernel<<<(H_SZ * H_SZ + 255) / 256, 256, 0, stream>>>(p3_W, p3_bf, H_SZ * H_SZ);

    init_kernel<<<(B_SZ * H_SZ + 255) / 256, 256, 0, stream>>>(
        h_in, gru_bi, gru_bh, h_f32, h_bf, bias_c);

    // 2) Gi = xn @ Wi^T + (bi + bh)   [131072 x 768, K=128]
    gemm_bf16_wmma<true, false, false>
        <<<dim3(T_LEN * B_SZ / 128, H3 / 128), 256, 0, stream>>>(
            xn_bf, Wi_bf, bias_c, Gi, T_LEN * B_SZ, H3, I_SZ);

    // 3) Serial GRU recurrence: Gh = h @ Wh^T, then fused gates.
    for (int t = 0; t < T_LEN; ++t) {
        gemm_bf16_wmma<false, false, false>
            <<<dim3(B_SZ / 128, H3 / 128), 256, 0, stream>>>(
                h_bf, Wh_bf, nullptr, Gh, B_SZ, H3, H_SZ);
        gru_gate_kernel<<<(B_SZ * H_SZ + 255) / 256, 256, 0, stream>>>(
            t, Gi, Gh, h_f32, h_bf, outs_bf, hlast_out);
    }

    // 4) MLP head (ReLU): two bf16 intermediates, final layer f32 -> d_out.
    gemm_bf16_wmma<true, true, true>
        <<<dim3(T_LEN * B_SZ / 128, H_SZ / 128), 256, 0, stream>>>(
            outs_bf, p1_bf, p1_b, y1_bf, T_LEN * B_SZ, H_SZ, H_SZ);
    gemm_bf16_wmma<true, true, true>
        <<<dim3(T_LEN * B_SZ / 128, H_SZ / 128), 256, 0, stream>>>(
            y1_bf, p2_bf, p2_b, y2_bf, T_LEN * B_SZ, H_SZ, H_SZ);
    gemm_bf16_wmma<true, true, false>
        <<<dim3(T_LEN * B_SZ / 128, H_SZ / 128), 256, 0, stream>>>(
            y2_bf, p3_bf, p3_b, y_out, T_LEN * B_SZ, H_SZ, H_SZ);
}
